// VLALayer_59725815218640
// MI455X (gfx1250) — compile-verified, matrix operand order checked by reading, not run on hardware
//
#include <hip/hip_runtime.h>
#include <math.h>

typedef __attribute__((ext_vector_type(16))) _Float16 v16h;
typedef __attribute__((ext_vector_type(4)))  _Float16 v4h;
typedef __attribute__((ext_vector_type(8)))  float    v8f;

#define D   512
#define TT  2048
#define BB  8
#define MROWS (BB*TT)

// ---------------------------------------------------------------------------
// GEMM: Out[m][n] = sum_k X[m][k] * W[n][k] + bias[n]   (x @ W.T + b)
// 128x128 tile / 256 threads / 8 waves. LDS tiles are stored in WMMA
// *fragment order* so each fragment load is 2x ds_load_b128. Global staging
// is float4 (global_load_b128), double-buffered to pipeline behind the WMMAs.
// ---------------------------------------------------------------------------
__global__ __launch_bounds__(256) void gemm_xwt_kernel(const float* __restrict__ X,
                                                       const float* __restrict__ W,
                                                       const float* __restrict__ bias,
                                                       float* __restrict__ Out)
{
    // fragment-ordered: [8 tiles][32 lanes][16 halfs]
    __shared__ __align__(32) _Float16 sA[2][4096];
    __shared__ __align__(32) _Float16 sB[2][4096];

    const int tid  = threadIdx.x;
    const int lane = tid & 31;
    const int wid  = tid >> 5;
    const int m0   = blockIdx.x * 128;
    const int n0   = blockIdx.y * 128;
    const int wm0  = wid * 16;

    v8f acc[8];
    #pragma unroll
    for (int i = 0; i < 8; ++i) acc[i] = (v8f){0.f,0.f,0.f,0.f,0.f,0.f,0.f,0.f};

    // Load one 128x32 (A) and 32x128 (B) chunk into registers (float4 each).
    auto load_chunk = [&](int k0, float4* ra, float4* rb) {
        #pragma unroll
        for (int j = 0; j < 4; ++j) {
            const int f = (tid + j * 256) * 4;     // linear f32 index in tile
            const int r = f >> 5, c = f & 31;      // A: row r (0..127), k c (0..31)
            ra[j] = *(const float4*)&X[(size_t)(m0 + r) * D + (k0 + c)];
            // B: n-major so global reads run along W's k (contiguous)
            rb[j] = *(const float4*)&W[(size_t)(n0 + r) * D + (k0 + c)];
        }
    };
    // Store registers into LDS in WMMA fragment order.
    //  (m|n, k) -> tile = (m|n)>>4, lane = (m|n&15) + 16*((k>>3)&1),
    //  elem i = (k&7) + 8*(k>>4)  (lo lanes: K 0-7/16-23, hi lanes: K 8-15/24-31)
    auto store_chunk = [&](int p, const float4* ra, const float4* rb) {
        #pragma unroll
        for (int j = 0; j < 4; ++j) {
            const int f = (tid + j * 256) * 4;
            const int r = f >> 5, c = f & 31;      // c is a multiple of 4
            const int dst = (((r >> 4) * 32 + ((c >> 3) & 1) * 16 + (r & 15)) << 4)
                            + (c & 7) + ((c >> 4) << 3);
            v4h ha = { (_Float16)ra[j].x, (_Float16)ra[j].y,
                       (_Float16)ra[j].z, (_Float16)ra[j].w };
            v4h hb = { (_Float16)rb[j].x, (_Float16)rb[j].y,
                       (_Float16)rb[j].z, (_Float16)rb[j].w };
            *(v4h*)&sA[p][dst] = ha;
            *(v4h*)&sB[p][dst] = hb;
        }
    };

    float4 ra[4], rb[4];
    load_chunk(0, ra, rb);
    store_chunk(0, ra, rb);
    __syncthreads();

    for (int kc = 0; kc < 16; ++kc) {
        const int p = kc & 1;
        float4 na[4], nb[4];
        if (kc + 1 < 16) load_chunk((kc + 1) * 32, na, nb);
        if (kc + 2 < 16) {  // warm L2/WGP$ for the chunk after next
            const int k2 = (kc + 2) * 32;
            __builtin_prefetch(&X[(size_t)(m0 + (tid & 127)) * D + k2], 0, 1);
            __builtin_prefetch(&W[(size_t)(n0 + (tid & 127)) * D + k2], 0, 1);
        }

        const v16h afrag = *(const v16h*)&sA[p][(wid * 32 + lane) << 4];
        #pragma unroll
        for (int nt = 0; nt < 8; ++nt) {
            const v16h bfrag = *(const v16h*)&sB[p][(nt * 32 + lane) << 4];
            acc[nt] = __builtin_amdgcn_wmma_f32_16x16x32_f16(
                false, afrag, false, bfrag, (short)0, acc[nt], false, false);
        }

        if (kc + 1 < 16) {
            store_chunk(1 - p, na, nb);   // other buffer: no hazard with reads of buf p
            __syncthreads();
        }
    }

    // C/D layout: VGPR j -> M = j + 8*(lane>=16); lane&15 -> N
    const int mrow_off = 8 * (lane >> 4);
    const int ncol     = lane & 15;
    for (int nt = 0; nt < 8; ++nt) {
        #pragma unroll
        for (int j = 0; j < 8; ++j) {
            const int m = m0 + wm0 + j + mrow_off;
            const int n = n0 + nt * 16 + ncol;
            Out[(size_t)m * D + n] = acc[nt][j] + bias[n];
        }
    }
}

// ---------------------------------------------------------------------------
// prep: s_seq[row] = sum(K*Q) ; V <- V / (||V|| + 1e-6)   (one wave per row)
// ---------------------------------------------------------------------------
__global__ __launch_bounds__(256) void prep_kernel(const float* __restrict__ Q,
                                                   const float* __restrict__ K,
                                                   float* __restrict__ V,
                                                   float* __restrict__ s_seq)
{
    const int lane = threadIdx.x & 31;
    const int wid  = threadIdx.x >> 5;
    const int row  = blockIdx.x * 8 + wid;
    const float4* q = (const float4*)(Q + (size_t)row * D);
    const float4* k = (const float4*)(K + (size_t)row * D);
    float4*       v = (float4*)(V + (size_t)row * D);

    float kq = 0.f, vs = 0.f;
    for (int i = lane; i < D / 4; i += 32) {
        float4 kk = k[i], qq = q[i], vv = v[i];
        kq += kk.x * qq.x + kk.y * qq.y + kk.z * qq.z + kk.w * qq.w;
        vs += vv.x * vv.x + vv.y * vv.y + vv.z * vv.z + vv.w * vv.w;
    }
    for (int off = 16; off > 0; off >>= 1) {
        kq += __shfl_xor(kq, off, 32);
        vs += __shfl_xor(vs, off, 32);
    }
    if (lane == 0) s_seq[row] = kq;
    const float scale = 1.0f / (sqrtf(vs) + 1e-6f);
    for (int i = lane; i < D / 4; i += 32) {
        float4 vv = v[i];
        vv.x *= scale; vv.y *= scale; vv.z *= scale; vv.w *= scale;
        v[i] = vv;
    }
}

__device__ inline void diag_add(float4& a, int comp, float e) {
    if      (comp == 0) a.x += e;
    else if (comp == 1) a.y += e;
    else if (comp == 2) a.z += e;
    else                a.w += e;
}

// ---------------------------------------------------------------------------
// Sequential scan: one workgroup (1024 threads) per batch; A,S L2-resident.
// Thread t owns row r=t>>1, column half t&1; all passes use float4.
// ---------------------------------------------------------------------------
__global__ __launch_bounds__(1024) void scan_kernel(const float* __restrict__ Q,
                                                    const float* __restrict__ K,
                                                    const float* __restrict__ Vn,
                                                    const float* __restrict__ s_seq,
                                                    float* __restrict__ Aall,
                                                    float* __restrict__ Sall,
                                                    float* __restrict__ O)
{
    __shared__ __align__(16) float su[D], sq[D], svn[D], sz[D], salpha[D];
    __shared__ float zp[1024];
    __shared__ float red[1024];

    const int tid = threadIdx.x;
    const int b   = blockIdx.x;
    float* A = Aall + (size_t)b * D * D;
    float* S = Sall + (size_t)b * D * D;

    for (int i = tid; i < D * D; i += 1024) {       // A = I/lambda_0, S = 0
        A[i] = (i % (D + 1) == 0) ? 1.0f : 0.0f;
        S[i] = 0.0f;
    }
    __syncthreads();

    const int   r  = tid >> 1;
    const int   c0 = (tid & 1) * 256;
    const int   dj = (r >= c0 && r < c0 + 256) ? ((r - c0) >> 2) : -1; // float4 holding diag
    const int   dc = r & 3;
    const float inv_sqrt_d = 0.044194173824159216f; // sqrt(lambda)=1, /sqrt(512)
    float eps_pre = 0.0f;

    for (int t = 0; t < TT; ++t) {
        const size_t base = ((size_t)b * TT + t) * D;
        for (int i = tid; i < D; i += 1024) {
            su[i]  = K[base + i] * inv_sqrt_d;
            sq[i]  = Q[base + i];
            svn[i] = Vn[base + i];
        }
        const float s_t = s_seq[(size_t)b * TT + t];
        __syncthreads();

        // ---- z = (A + eps_pre*I) u ----
        {
            float acc = 0.f;
            const float4* Ar4 = (const float4*)(A + (size_t)r * D + c0);
            const float4* u4  = (const float4*)(su + c0);
            for (int j = 0; j < 64; ++j) {
                float4 a = Ar4[j];
                if (j == dj) diag_add(a, dc, eps_pre);
                const float4 u = u4[j];
                acc += a.x * u.x + a.y * u.y + a.z * u.z + a.w * u.w;
            }
            zp[tid] = acc;
        }
        __syncthreads();
        if (tid < D) {
            const float zr = zp[2 * tid] + zp[2 * tid + 1];
            sz[tid]  = zr;
            red[tid] = zr * su[tid];
        } else red[tid] = 0.f;
        __syncthreads();
        for (int s = 512; s > 0; s >>= 1) {
            if (tid < s) red[tid] += red[tid + s];
            __syncthreads();
        }
        const float dot   = red[0];
        const float dd    = 1.0f + dot;
        const float delta = (dd > 1e-6f) ? dd : ((dd == dd) ? 1e-6f : dd); // max w/ NaN-prop
        const bool  unstable  = (fabsf(delta) < 1e-6f) || !__builtin_isfinite(delta);
        const float inv_delta = 1.0f / delta;
        const float eps_mid   = (((t + 1) & 127) == 0) ? 1e-6f : 0.0f;
        __syncthreads();

        // ---- A update (Sherman-Morrison) fused with z2 = A_new u and ||A||^2 ----
        {
            float z2acc = 0.f, asum = 0.f;
            float4* Ar4 = (float4*)(A + (size_t)r * D + c0);
            const float4* u4 = (const float4*)(su + c0);
            const float4* z4 = (const float4*)(sz + c0);
            const float zr = sz[r];
            const float diag_eps = (unstable ? 1e-6f : 0.0f) + eps_mid;
            for (int j = 0; j < 64; ++j) {
                float4 a = Ar4[j];
                if (j == dj) diag_add(a, dc, eps_pre);
                float4 nv;
                if (unstable) {
                    nv = a;
                } else {
                    const float4 zc = z4[j];
                    nv.x = a.x - zr * zc.x * inv_delta;
                    nv.y = a.y - zr * zc.y * inv_delta;
                    nv.z = a.z - zr * zc.z * inv_delta;
                    nv.w = a.w - zr * zc.w * inv_delta;
                }
                if (j == dj) diag_add(nv, dc, diag_eps);
                Ar4[j] = nv;
                const float4 u = u4[j];
                z2acc += nv.x * u.x + nv.y * u.y + nv.z * u.z + nv.w * u.w;
                asum  += nv.x * nv.x + nv.y * nv.y + nv.z * nv.z + nv.w * nv.w;
            }
            zp[tid]  = z2acc;
            red[tid] = asum;
        }
        __syncthreads();
        if (tid < D) salpha[tid] = s_t * (zp[2 * tid] + zp[2 * tid + 1]);
        for (int s = 512; s > 0; s >>= 1) {
            if (tid < s) red[tid] += red[tid + s];
            __syncthreads();
        }
        const float Asumsq = red[0];
        __syncthreads();

        // ---- S += vn alpha^T fused with o = S_new q and ||S||^2 ----
        {
            float oacc = 0.f, ssum = 0.f;
            float4* Sr4 = (float4*)(S + (size_t)r * D + c0);
            const float4* a4 = (const float4*)(salpha + c0);
            const float4* q4 = (const float4*)(sq + c0);
            const float vnr = svn[r];
            for (int j = 0; j < 64; ++j) {
                float4 sv = Sr4[j];
                const float4 al = a4[j];
                sv.x += vnr * al.x; sv.y += vnr * al.y;
                sv.z += vnr * al.z; sv.w += vnr * al.w;
                Sr4[j] = sv;
                const float4 q = q4[j];
                oacc += sv.x * q.x + sv.y * q.y + sv.z * q.z + sv.w * q.w;
                ssum += sv.x * sv.x + sv.y * sv.y + sv.z * sv.z + sv.w * sv.w;
            }
            zp[tid]  = oacc;
            red[tid] = ssum;
        }
        __syncthreads();
        if (tid < D) O[base + tid] = zp[2 * tid] + zp[2 * tid + 1];
        for (int s = 512; s > 0; s >>= 1) {
            if (tid < s) red[tid] += red[tid + s];
            __syncthreads();
        }
        const float Ssumsq = red[0];
        // norm > 1000 <=> sumsq > 1e6 (NaN compares false, matching jnp)
        eps_pre = ((Ssumsq > 1.0e6f) || (Asumsq > 1.0e6f)) ? 1e-5f : 0.0f;
        __syncthreads();
    }
}

// ---------------------------------------------------------------------------
extern "C" void kernel_launch(void* const* d_in, const int* in_sizes, int n_in,
                              void* d_out, int out_size, void* d_ws, size_t ws_size,
                              hipStream_t stream)
{
    (void)in_sizes; (void)n_in; (void)out_size; (void)ws_size;
    const float* x  = (const float*)d_in[0];
    const float* Wq = (const float*)d_in[1];
    const float* bq = (const float*)d_in[2];
    const float* Wk = (const float*)d_in[3];
    const float* bk = (const float*)d_in[4];
    const float* Wv = (const float*)d_in[5];
    const float* bv = (const float*)d_in[6];
    const float* Wo = (const float*)d_in[7];
    const float* bo = (const float*)d_in[8];
    float* out = (float*)d_out;

    // workspace layout (floats): Q | K | V(->Vn) | O | s_seq | A[8] | S[8]
    float* ws = (float*)d_ws;
    float* Qf = ws;
    float* Kf = Qf + (size_t)MROWS * D;
    float* Vf = Kf + (size_t)MROWS * D;
    float* Of = Vf + (size_t)MROWS * D;
    float* sf = Of + (size_t)MROWS * D;
    float* Af = sf + MROWS;
    float* Sf = Af + (size_t)BB * D * D;

    dim3 gg(MROWS / 128, D / 128);
    gemm_xwt_kernel<<<gg, 256, 0, stream>>>(x, Wq, bq, Qf);
    gemm_xwt_kernel<<<gg, 256, 0, stream>>>(x, Wk, bk, Kf);
    gemm_xwt_kernel<<<gg, 256, 0, stream>>>(x, Wv, bv, Vf);
    prep_kernel<<<MROWS / 8, 256, 0, stream>>>(Qf, Kf, Vf, sf);
    scan_kernel<<<BB, 1024, 0, stream>>>(Qf, Kf, Vf, sf, Af, Sf, Of);
    gemm_xwt_kernel<<<gg, 256, 0, stream>>>(Of, Wo, bo, out);
}
